// SATLayer_46866683134522
// MI455X (gfx1250) — compile-verified, matrix operand order checked by reading, not run on hardware
//
#include <hip/hip_runtime.h>
#include <hip/hip_bf16.h>

// ---- problem constants (match reference) ----
#define NN    100000
#define EE    1600000
#define K_IN  256
#define N_OUT 128
#define M_TILES (NN / 16)        // 6250, exact
#define M_PAIRS (M_TILES / 2)    // 3125, exact (32 rows per wave)

typedef __attribute__((ext_vector_type(16))) __bf16 v16bf;
typedef __attribute__((ext_vector_type(8)))  float  v8f;
typedef __attribute__((ext_vector_type(4)))  float  v4f;

// ------------------------------------------------------------------
// K0: init rowmax (encoded -inf) and rowsum (0)
// ------------------------------------------------------------------
__global__ void init_seg_kernel(unsigned* __restrict__ rowmax,
                                float* __restrict__ rowsum) {
    int i = blockIdx.x * 256 + threadIdx.x;
    if (i < NN) {
        rowmax[i] = 0x007FFFFFu;   // ordered-uint encoding of -inf
        rowsum[i] = 0.0f;
    }
}

// ------------------------------------------------------------------
// A-fragment loader: row-major f32 -> WMMA bf16 A layout.
// lanes 0-15 hold K {k0..k0+7, k0+16..k0+23}; lanes 16-31 get k0+=8.
// ------------------------------------------------------------------
__device__ __forceinline__ v16bf load_a_frag(const float* __restrict__ arow, int k0) {
    v4f f0 = *(const v4f*)(arow + k0);
    v4f f1 = *(const v4f*)(arow + k0 + 4);
    v4f f2 = *(const v4f*)(arow + k0 + 16);
    v4f f3 = *(const v4f*)(arow + k0 + 20);
    v16bf a;
    #pragma unroll
    for (int j = 0; j < 4; ++j) {
        a[j]      = (__bf16)f0[j];
        a[4 + j]  = (__bf16)f1[j];
        a[8 + j]  = (__bf16)f2[j];
        a[12 + j] = (__bf16)f3[j];
    }
    return a;
}

// ------------------------------------------------------------------
// K1: h = features @ W + b   via bf16 WMMA (f32 accumulate)
// One wave computes a 32x128 tile of h (two 16-row WMMA tiles sharing
// each B fragment). W staged in LDS in a fragment-shuffled bf16 layout:
// frag(kt,nt) x lane -> 16 contiguous bf16 (32 B per lane).
// ------------------------------------------------------------------
__global__ __launch_bounds__(256) void gemm_wmma_kernel(
    const float* __restrict__ feat,   // [NN, K_IN]
    const float* __restrict__ W,      // [K_IN, N_OUT]
    const float* __restrict__ bias,   // [N_OUT]
    float* __restrict__ h)            // [NN, N_OUT]
{
    // 64 fragments (8 ktiles x 8 ntiles) * 32 lanes * 16 bf16 = 64 KB
    __shared__ __bf16 ldsB[64 * 32 * 16];

    const int t = threadIdx.x;
    // cooperative fill of B fragments (f32 -> bf16, WMMA B layout)
    for (int i = t; i < K_IN * N_OUT; i += 256) {
        int k = i / N_OUT, n = i % N_OUT;
        int ktile = k >> 5, kk = k & 31;
        int ntile = n >> 4, nn = n & 15;
        int laneHalf = (kk >> 3) & 1;            // K 8-15 / 24-31 -> lanes 16-31
        int lane = nn + (laneHalf << 4);
        int idx = (kk & 7) | ((kk & 16) >> 1);   // position within lane's 16 elems
        ldsB[(((ktile * 8 + ntile) * 32 + lane) << 4) + idx] = (__bf16)W[i];
    }
    __syncthreads();

    const int wave     = t >> 5;
    const int lane     = t & 31;
    const int laneLo   = lane & 15;
    const int laneHalf = lane >> 4;

    for (int pair = blockIdx.x * 8 + wave; pair < M_PAIRS; pair += gridDim.x * 8) {
        const int rowBase0 = pair * 32;
        const int rowBase1 = rowBase0 + 16;

        v8f acc0[8], acc1[8];
        #pragma unroll
        for (int nt = 0; nt < 8; ++nt) {
            acc0[nt] = (v8f){0.f, 0.f, 0.f, 0.f, 0.f, 0.f, 0.f, 0.f};
            acc1[nt] = (v8f){0.f, 0.f, 0.f, 0.f, 0.f, 0.f, 0.f, 0.f};
        }

        const float* arow0 = feat + (size_t)(rowBase0 + laneLo) * K_IN;
        const float* arow1 = feat + (size_t)(rowBase1 + laneLo) * K_IN;

        #pragma unroll
        for (int kt = 0; kt < 8; ++kt) {
            // Opaque zero offset: defeats LICM so B-fragment ds_loads stay
            // inside the loop instead of being hoisted (and spilled).
            unsigned ldsOff = 0;
            asm volatile("" : "+v"(ldsOff));

            const int k0 = kt * 32 + (laneHalf << 3);
            v16bf aF0 = load_a_frag(arow0, k0);
            v16bf aF1 = load_a_frag(arow1, k0);

            #pragma unroll
            for (int nt = 0; nt < 8; ++nt) {
                v16bf bFrag = *(const v16bf*)&ldsB[ldsOff + (((kt * 8 + nt) * 32 + lane) << 4)];
                acc0[nt] = __builtin_amdgcn_wmma_f32_16x16x32_bf16(
                    false, aF0, false, bFrag, (short)0, acc0[nt], false, false);
                acc1[nt] = __builtin_amdgcn_wmma_f32_16x16x32_bf16(
                    false, aF1, false, bFrag, (short)0, acc1[nt], false, false);
            }
        }

        // D layout: VGPR r -> M = rowBase + r + 8*laneHalf, N = nt*16 + laneLo
        #pragma unroll
        for (int nt = 0; nt < 8; ++nt) {
            const int col = nt * 16 + laneLo;
            const float bv = bias[col];
            #pragma unroll
            for (int r = 0; r < 8; ++r) {
                const int m0 = rowBase0 + r + (laneHalf << 3);
                const int m1 = rowBase1 + r + (laneHalf << 3);
                h[(size_t)m0 * N_OUT + col] = acc0[nt][r] + bv;
                h[(size_t)m1 * N_OUT + col] = acc1[nt][r] + bv;
            }
        }
    }
}

// ------------------------------------------------------------------
// K2: a1[i] = sum_j |h[i,j]|*a1_w[j] + a1_b ; same for a2. One wave/row.
// ------------------------------------------------------------------
__global__ __launch_bounds__(256) void attn_vec_kernel(
    const float* __restrict__ h,
    const float* __restrict__ a1w, const float* __restrict__ a1b,
    const float* __restrict__ a2w, const float* __restrict__ a2b,
    float* __restrict__ a1, float* __restrict__ a2)
{
    const int wave = threadIdx.x >> 5;
    const int lane = threadIdx.x & 31;
    const int row  = blockIdx.x * 8 + wave;
    if (row >= NN) return;

    const float* hr = h + (size_t)row * N_OUT;
    float s1 = 0.f, s2 = 0.f;
    #pragma unroll
    for (int j = lane; j < N_OUT; j += 32) {
        float v = fabsf(hr[j]);
        s1 += v * a1w[j];
        s2 += v * a2w[j];
    }
    #pragma unroll
    for (int off = 16; off >= 1; off >>= 1) {
        s1 += __shfl_xor(s1, off, 32);
        s2 += __shfl_xor(s2, off, 32);
    }
    if (lane == 0) {
        a1[row] = s1 + a1b[0];
        a2[row] = s2 + a2b[0];
    }
}

// ------------------------------------------------------------------
// K3: logits + segment max (ordered-uint atomicMax)
// ------------------------------------------------------------------
__global__ void logits_max_kernel(const float* __restrict__ a1,
                                  const float* __restrict__ a2,
                                  const int* __restrict__ row,
                                  const int* __restrict__ col,
                                  float* __restrict__ logits,
                                  unsigned* __restrict__ rowmax)
{
    int e = blockIdx.x * 256 + threadIdx.x;
    if (e >= EE) return;
    int r = row[e], c = col[e];
    float lg = a1[r] + a2[c];
    logits[e] = lg;
    unsigned u = __float_as_uint(lg);
    u = (u & 0x80000000u) ? ~u : (u | 0x80000000u);
    atomicMax(rowmax + r, u);
}

// K3b: decode ordered-uint max back to float (in place)
__global__ void decode_max_kernel(unsigned* __restrict__ rowmax) {
    int i = blockIdx.x * 256 + threadIdx.x;
    if (i < NN) {
        unsigned u = rowmax[i];
        u = (u & 0x80000000u) ? (u & 0x7FFFFFFFu) : ~u;
        rowmax[i] = u;   // bits now hold the float value
    }
}

// ------------------------------------------------------------------
// K4: ex = exp(logit - max[row]); segment sum. Overwrites logits with ex.
// ------------------------------------------------------------------
__global__ void expsum_kernel(float* __restrict__ logits_ex,
                              const int* __restrict__ row,
                              const float* __restrict__ rowmax,
                              float* __restrict__ rowsum)
{
    int e = blockIdx.x * 256 + threadIdx.x;
    if (e >= EE) return;
    int r = row[e];
    float ex = expf(logits_ex[e] - rowmax[r]);
    logits_ex[e] = ex;
    atomicAdd(rowsum + r, ex);
}

// ------------------------------------------------------------------
// K5: out[row] += (ex/rowsum[row]) * values[e] * h[col].  One wave/edge,
// each lane owns a float4 column chunk (128 = 32 lanes * 4).
// ------------------------------------------------------------------
__global__ __launch_bounds__(256) void spmm_kernel(
    const float* __restrict__ ex,
    const int* __restrict__ row, const int* __restrict__ col,
    const float* __restrict__ values,
    const float* __restrict__ rowsum,
    const float* __restrict__ h,
    float* __restrict__ out)
{
    long long gid = (long long)blockIdx.x * 256 + threadIdx.x;
    int e    = (int)(gid >> 5);
    int lane = (int)(gid & 31);
    if (e >= EE) return;
    int r = row[e], c = col[e];
    float av = (ex[e] / rowsum[r]) * values[e];
    v4f hv = *(const v4f*)(h + (size_t)c * N_OUT + lane * 4);
    float* op = out + (size_t)r * N_OUT + lane * 4;
    atomicAdd(op + 0, av * hv.x);
    atomicAdd(op + 1, av * hv.y);
    atomicAdd(op + 2, av * hv.z);
    atomicAdd(op + 3, av * hv.w);
}

// ------------------------------------------------------------------
extern "C" void kernel_launch(void* const* d_in, const int* in_sizes, int n_in,
                              void* d_out, int out_size, void* d_ws, size_t ws_size,
                              hipStream_t stream)
{
    const float* feat   = (const float*)d_in[0];   // [NN, K_IN]
    const float* values = (const float*)d_in[1];   // [EE]
    const float* W      = (const float*)d_in[2];   // [K_IN, N_OUT]
    const float* bias   = (const float*)d_in[3];   // [N_OUT]
    const float* a1w    = (const float*)d_in[4];   // [N_OUT]
    const float* a1b    = (const float*)d_in[5];   // [1]
    const float* a2w    = (const float*)d_in[6];   // [N_OUT]
    const float* a2b    = (const float*)d_in[7];   // [1]
    const int*   row    = (const int*)d_in[8];     // [EE]
    const int*   col    = (const int*)d_in[9];     // [EE]
    float* out = (float*)d_out;                    // [NN, N_OUT]

    // workspace layout (256B-aligned slabs)
    char* ws = (char*)d_ws;
    size_t off = 0;
    auto alloc = [&](size_t bytes) {
        void* p = ws + off;
        off += (bytes + 255) & ~(size_t)255;
        return p;
    };
    float*    h      = (float*)alloc((size_t)NN * N_OUT * sizeof(float));
    float*    a1     = (float*)alloc((size_t)NN * sizeof(float));
    float*    a2     = (float*)alloc((size_t)NN * sizeof(float));
    unsigned* rowmax = (unsigned*)alloc((size_t)NN * sizeof(unsigned));
    float*    rowsum = (float*)alloc((size_t)NN * sizeof(float));
    float*    logits = (float*)alloc((size_t)EE * sizeof(float));   // becomes ex

    (void)in_sizes; (void)n_in; (void)out_size; (void)ws_size;

    // zero output accumulator
    hipMemsetAsync(out, 0, (size_t)NN * N_OUT * sizeof(float), stream);

    // K0
    init_seg_kernel<<<(NN + 255) / 256, 256, 0, stream>>>(rowmax, rowsum);

    // K1: GEMM with bf16 WMMA (32 rows per wave)
    gemm_wmma_kernel<<<(M_PAIRS + 7) / 8, 256, 0, stream>>>(feat, W, bias, h);

    // K2: a1/a2
    attn_vec_kernel<<<(NN + 7) / 8, 256, 0, stream>>>(h, a1w, a1b, a2w, a2b, a1, a2);

    // K3: logits + segment max
    logits_max_kernel<<<(EE + 255) / 256, 256, 0, stream>>>(a1, a2, row, col, logits, rowmax);
    decode_max_kernel<<<(NN + 255) / 256, 256, 0, stream>>>(rowmax);

    // K4: exp + segment sum
    expsum_kernel<<<(EE + 255) / 256, 256, 0, stream>>>(logits, row, (const float*)rowmax, rowsum);

    // K5: attention-weighted SpMM
    long long threads = (long long)EE * 32;
    spmm_kernel<<<(unsigned)((threads + 255) / 256), 256, 0, stream>>>(
        logits, row, col, values, rowsum, h, out);
}
